// Net_60052232733094
// MI455X (gfx1250) — matrix operand order repaired and checked by reading, not compiled, not run on hardware
//
#include <hip/hip_runtime.h>
#include <hip/hip_bf16.h>

typedef __attribute__((ext_vector_type(16))) _Float16 v16h;
typedef __attribute__((ext_vector_type(8)))  float    v8f;
typedef __attribute__((ext_vector_type(4)))  float    f32x4;

// ---------------- sizes ----------------
#define N1 524288
#define N2 131072
#define N3 32768
#define E1 4194304
#define E2 1048576
#define E3 262144
#define SEGS 16384

// ---------------- workspace layout (bytes) ----------------
#define OFF_ACC  ((size_t)0)
#define OFF_H1   ((size_t)67108864)    /* inside ACC; acc1 = N1*32*4 = 67,108,864 B ends here */
#define OFF_H3   ((size_t)268435456)   /* inside ACC; acc3 = 209,715,200 B < 256MiB */
#define OFF_DEG  ((size_t)419430400)
#define OFF_P1   ((size_t)421527552)
#define OFF_H2   ((size_t)438304768)
#define OFF_P2   ((size_t)471859200)
#define OFF_P3   ((size_t)480247808)
#define OFF_Z1   ((size_t)488636416)
#define OFF_W1F  ((size_t)490733568)
#define OFF_W2F  ((size_t)490737664)
#define OFF_R2F  ((size_t)490840064)
#define OFF_W3F  ((size_t)490844160)
#define OFF_R3F  ((size_t)491253760)
#define OFF_FCF  ((size_t)491270144)

#define ENC_NEGINF 0x007FFFFFu  /* monotonic encoding of -inf */

// Hardware f32 atomic add (non-returning) — avoid CAS fallback.
__device__ __forceinline__ void atomicAddF32(float* p, float v) {
    asm volatile("global_atomic_add_f32 %0, %1, off" :: "v"(p), "v"(v) : "memory");
}

// ---------------- small utility kernels ----------------
__global__ void fill_u32_kernel(unsigned* p, unsigned val, long n) {
    long i = (long)blockIdx.x * blockDim.x + threadIdx.x;
    if (i < n) p[i] = val;
}

// Swizzle f32 weights [K][N] into fragment-major f16 blocks:
// dst[((kb*TN + t)*32 + lane)*16 + e] = W[kb*32 + (lane>>4)*8 + (e&7) + (e>>3)*16][t*16 + (lane&15)]
// zero-padded for k >= K.  TN = N/16, total = Kpad*N elements.
__global__ void swizzle_b_kernel(const float* __restrict__ src, _Float16* __restrict__ dst,
                                 int K, int Kpad, int N) {
    long total = (long)Kpad * N;
    long i = (long)blockIdx.x * blockDim.x + threadIdx.x;
    if (i >= total) return;
    int e = (int)(i & 15);
    int lane = (int)((i >> 4) & 31);
    long blk = i >> 9;                 // kb*TN + t
    int tn = N >> 4;
    int kb = (int)(blk / tn);
    int t  = (int)(blk - (long)kb * tn);
    int k   = kb * 32 + ((lane >> 4) * 8) + (e & 7) + ((e >> 3) * 16);
    int col = t * 16 + (lane & 15);
    float v = (k < K) ? src[(size_t)k * N + col] : 0.0f;
    dst[i] = (_Float16)v;
}

// ---------------- edge scatter (B-spline corner accumulation) ----------------
// acc layout [Nnodes][node_stride]; one thread per (edge, channel), Cin = 1<<cshift.
__global__ void scatter_edges_kernel(const int* __restrict__ ei,
                                     const float* __restrict__ pseudo,
                                     const float* __restrict__ xsrc,
                                     float* __restrict__ acc,
                                     float* __restrict__ deg,
                                     int E, int cshift, int node_stride) {
    long tid = (long)blockIdx.x * blockDim.x + threadIdx.x;
    long e = tid >> cshift;
    if (e >= (long)E) return;
    int c = (int)(tid & ((1 << cshift) - 1));
    int s = ei[e];
    int d = ei[(long)E + e];
    float px = pseudo[2 * e] * 4.0f;
    float py = pseudo[2 * e + 1] * 4.0f;
    float lxf = fminf(fmaxf(floorf(px), 0.0f), 3.0f);
    float lyf = fminf(fmaxf(floorf(py), 0.0f), 3.0f);
    int lx = (int)lxf, ly = (int)lyf;
    float fx = px - lxf, fy = py - lyf;
    float v = xsrc[((size_t)s << cshift) + c];
    float* base = acc + (size_t)d * node_stride + c;
    float w00 = (1.0f - fx) * (1.0f - fy);
    float w01 = (1.0f - fx) * fy;
    float w10 = fx * (1.0f - fy);
    float w11 = fx * fy;
    atomicAddF32(base + ((size_t)(lx * 5 + ly)           << cshift), w00 * v);
    atomicAddF32(base + ((size_t)(lx * 5 + ly + 1)       << cshift), w01 * v);
    atomicAddF32(base + ((size_t)((lx + 1) * 5 + ly)     << cshift), w10 * v);
    atomicAddF32(base + ((size_t)((lx + 1) * 5 + ly + 1) << cshift), w11 * v);
    if (c == 0) atomicAddF32(deg + d, 1.0f);
}

// acc[n][:] *= 1/max(deg[n],1)   (one block per node)
__global__ void normalize_acc_kernel(float* acc, const float* deg, int rowlen) {
    int n = blockIdx.x;
    float s = 1.0f / fmaxf(deg[n], 1.0f);
    float* row = acc + (size_t)n * rowlen;
    for (int i = threadIdx.x; i < rowlen; i += blockDim.x) row[i] *= s;
}

// ---------------- WMMA GEMM ----------------
// out[M][N] = elu?( A1[M][lda1](K1pad cols) * B1sw + A2 * B2sw + xvec·rootrow + bias )
// A* dense f32, K*pad multiple of 32, no bound checks (padding lives in memory).
// B*sw pre-swizzled f16 fragments.  One wave per 16-row strip, TN = N/16 tiles wide.
__device__ __forceinline__ v16h load_a_frag_nt(const float* arow, int k0, int h) {
    const f32x4* pa = (const f32x4*)(arow + k0 + 8 * h);
    f32x4 q0 = __builtin_nontemporal_load(pa + 0);
    f32x4 q1 = __builtin_nontemporal_load(pa + 1);
    f32x4 q2 = __builtin_nontemporal_load(pa + 4);
    f32x4 q3 = __builtin_nontemporal_load(pa + 5);
    v16h a;
    a[0]  = (_Float16)q0.x;  a[1]  = (_Float16)q0.y;
    a[2]  = (_Float16)q0.z;  a[3]  = (_Float16)q0.w;
    a[4]  = (_Float16)q1.x;  a[5]  = (_Float16)q1.y;
    a[6]  = (_Float16)q1.z;  a[7]  = (_Float16)q1.w;
    a[8]  = (_Float16)q2.x;  a[9]  = (_Float16)q2.y;
    a[10] = (_Float16)q2.z;  a[11] = (_Float16)q2.w;
    a[12] = (_Float16)q3.x;  a[13] = (_Float16)q3.y;
    a[14] = (_Float16)q3.z;  a[15] = (_Float16)q3.w;
    return a;
}

template <int TN>
__global__ __launch_bounds__(256) void wmma_gemm_kernel(
        const float* __restrict__ A1, int lda1, int K1pad,
        const _Float16* __restrict__ B1sw,
        const float* __restrict__ A2, int lda2, int K2pad,
        const _Float16* __restrict__ B2sw,
        const float* __restrict__ xvec, const float* __restrict__ rootrow,
        const float* __restrict__ bias, float* __restrict__ out,
        int M, int do_elu) {
    constexpr int N = TN * 16;
    int wid  = (int)(((long)blockIdx.x * blockDim.x + threadIdx.x) >> 5);
    int lane = (int)(threadIdx.x & 31);
    if (wid >= (M >> 4)) return;          // wave-uniform: EXEC stays all-ones
    int mt = wid;
    int r = lane & 15, h = lane >> 4;
    v8f cacc[TN] = {};
    {
        const float* arow = A1 + (size_t)(mt * 16 + r) * lda1;
        for (int k0 = 0; k0 < K1pad; k0 += 32) {
            v16h a = load_a_frag_nt(arow, k0, h);
            const _Float16* bb = B1sw + ((size_t)(k0 >> 5) * TN) * 512 + lane * 16;
#pragma unroll
            for (int t = 0; t < TN; ++t) {
                v16h b = *(const v16h*)(bb + (size_t)t * 512);
                cacc[t] = __builtin_amdgcn_wmma_f32_16x16x32_f16(
                    false, a, false, b, (short)0, cacc[t], false, false);
            }
        }
    }
    if (K2pad > 0) {
        const float* arow = A2 + (size_t)(mt * 16 + r) * lda2;
        for (int k0 = 0; k0 < K2pad; k0 += 32) {
            v16h a = load_a_frag_nt(arow, k0, h);
            const _Float16* bb = B2sw + ((size_t)(k0 >> 5) * TN) * 512 + lane * 16;
#pragma unroll
            for (int t = 0; t < TN; ++t) {
                v16h b = *(const v16h*)(bb + (size_t)t * 512);
                cacc[t] = __builtin_amdgcn_wmma_f32_16x16x32_f16(
                    false, a, false, b, (short)0, cacc[t], false, false);
            }
        }
    }
    float xv[8];
    if (xvec) {
#pragma unroll
        for (int v = 0; v < 8; ++v) xv[v] = xvec[mt * 16 + v + 8 * h];
    }
#pragma unroll
    for (int t = 0; t < TN; ++t) {
        int col = t * 16 + r;
        float bs = bias[col];
        float rv = xvec ? rootrow[col] : 0.0f;
#pragma unroll
        for (int v = 0; v < 8; ++v) {
            int m = mt * 16 + v + 8 * h;
            float val = cacc[t][v] + bs;
            if (xvec) val += xv[v] * rv;
            if (do_elu) val = (val > 0.0f) ? val : expm1f(val);
            out[(size_t)m * N + col] = val;
        }
    }
}

// ---------------- voxel max pool via monotonic-u32 atomicMax ----------------
__global__ void pool_scatter_kernel(const float* __restrict__ h,
                                    const int* __restrict__ cluster,
                                    unsigned* __restrict__ obits,
                                    long Nn, int cshift) {
    long tid = (long)blockIdx.x * blockDim.x + threadIdx.x;
    long n = tid >> cshift;
    if (n >= Nn) return;
    int c = (int)(tid & ((1 << cshift) - 1));
    float f = h[tid];
    unsigned bi = __float_as_uint(f);
    unsigned u = (bi & 0x80000000u) ? ~bi : (bi | 0x80000000u);
    atomicMax(&obits[((size_t)cluster[n] << cshift) + c], u);
}

__global__ void pool_decode_kernel(unsigned* buf, long n) {
    long i = (long)blockIdx.x * blockDim.x + threadIdx.x;
    if (i >= n) return;
    unsigned u = buf[i];
    float f;
    if (u == ENC_NEGINF) {
        f = 0.0f;                       // empty voxel -> 0
    } else {
        unsigned bi = (u & 0x80000000u) ? (u & 0x7FFFFFFFu) : ~u;
        f = __uint_as_float(bi);
    }
    ((float*)buf)[i] = f;
}

// ---------------- head: z@fc2_w + fc2_b, log_softmax ----------------
__global__ void head_kernel(const float* __restrict__ z1,
                            const float* __restrict__ w,
                            const float* __restrict__ b,
                            float* __restrict__ out) {
    int row = blockIdx.x * blockDim.x + threadIdx.x;
    if (row >= 4096) return;
    const float* z = z1 + (size_t)row * 128;
    float o[10];
#pragma unroll
    for (int j = 0; j < 10; ++j) o[j] = b[j];
    for (int k = 0; k < 128; ++k) {
        float zv = z[k];
        const float* wr = w + k * 10;
#pragma unroll
        for (int j = 0; j < 10; ++j) o[j] += zv * wr[j];
    }
    float m = o[0];
#pragma unroll
    for (int j = 1; j < 10; ++j) m = fmaxf(m, o[j]);
    float s = 0.0f;
#pragma unroll
    for (int j = 0; j < 10; ++j) s += expf(o[j] - m);
    float lse = m + logf(s);
#pragma unroll
    for (int j = 0; j < 10; ++j) out[(size_t)row * 10 + j] = o[j] - lse;
}

// ---------------- launch ----------------
static inline int blks(long threads) { return (int)((threads + 255) / 256); }

extern "C" void kernel_launch(void* const* d_in, const int* in_sizes, int n_in,
                              void* d_out, int out_size, void* d_ws, size_t ws_size,
                              hipStream_t stream) {
    (void)in_sizes; (void)n_in; (void)out_size; (void)ws_size;
    const float* x    = (const float*)d_in[0];
    const int*   ei1  = (const int*)  d_in[1];
    const float* ps1  = (const float*)d_in[2];
    const int*   cl1  = (const int*)  d_in[3];
    const int*   ei2  = (const int*)  d_in[4];
    const float* ps2  = (const float*)d_in[5];
    const int*   cl2  = (const int*)  d_in[6];
    const int*   ei3  = (const int*)  d_in[7];
    const float* ps3  = (const float*)d_in[8];
    const int*   cl3  = (const int*)  d_in[9];
    const float* W1   = (const float*)d_in[10];
    const float* r1   = (const float*)d_in[11];
    const float* b1   = (const float*)d_in[12];
    const float* W2   = (const float*)d_in[13];
    const float* r2   = (const float*)d_in[14];
    const float* b2   = (const float*)d_in[15];
    const float* W3   = (const float*)d_in[16];
    const float* r3   = (const float*)d_in[17];
    const float* b3   = (const float*)d_in[18];
    const float* fc1w = (const float*)d_in[19];
    const float* fc1b = (const float*)d_in[20];
    const float* fc2w = (const float*)d_in[21];
    const float* fc2b = (const float*)d_in[22];

    char* ws = (char*)d_ws;
    float*     acc  = (float*)(ws + OFF_ACC);
    float*     h1   = (float*)(ws + OFF_H1);
    float*     h3   = (float*)(ws + OFF_H3);
    float*     deg  = (float*)(ws + OFF_DEG);
    float*     p1   = (float*)(ws + OFF_P1);
    float*     h2   = (float*)(ws + OFF_H2);
    float*     p2   = (float*)(ws + OFF_P2);
    float*     p3   = (float*)(ws + OFF_P3);
    float*     z1   = (float*)(ws + OFF_Z1);
    _Float16*  w1f  = (_Float16*)(ws + OFF_W1F);
    _Float16*  w2f  = (_Float16*)(ws + OFF_W2F);
    _Float16*  r2f  = (_Float16*)(ws + OFF_R2F);
    _Float16*  w3f  = (_Float16*)(ws + OFF_W3F);
    _Float16*  r3f  = (_Float16*)(ws + OFF_R3F);
    _Float16*  fcf  = (_Float16*)(ws + OFF_FCF);

    // weight swizzle (f32 [K][N] -> fragment-major f16, zero k-padding)
    swizzle_b_kernel<<<blks(1024),   256, 0, stream>>>(W1,   w1f, 25,   32,   32);
    swizzle_b_kernel<<<blks(51200),  256, 0, stream>>>(W2,   w2f, 800,  800,  64);
    swizzle_b_kernel<<<blks(2048),   256, 0, stream>>>(r2,   r2f, 32,   32,   64);
    swizzle_b_kernel<<<blks(204800), 256, 0, stream>>>(W3,   w3f, 1600, 1600, 128);
    swizzle_b_kernel<<<blks(8192),   256, 0, stream>>>(r3,   r3f, 64,   64,   128);
    swizzle_b_kernel<<<blks(65536),  256, 0, stream>>>(fc1w, fcf, 512,  512,  128);

    // -------- layer 1 (Cin=1 -> 32); acc1 padded to [N1][32] --------
    hipMemsetAsync(acc, 0, (size_t)N1 * 32 * 4, stream);
    hipMemsetAsync(deg, 0, (size_t)N1 * 4, stream);
    scatter_edges_kernel<<<blks((long)E1), 256, 0, stream>>>(ei1, ps1, x, acc, deg, E1, 0, 32);
    normalize_acc_kernel<<<N1, 256, 0, stream>>>(acc, deg, 32);
    wmma_gemm_kernel<2><<<blks((long)(N1 / 16) * 32), 256, 0, stream>>>(
        acc, 32, 32, w1f, nullptr, 0, 0, nullptr, x, r1, b1, h1, N1, 1);
    fill_u32_kernel<<<blks((long)N2 * 32), 256, 0, stream>>>((unsigned*)p1, ENC_NEGINF, (long)N2 * 32);
    pool_scatter_kernel<<<blks((long)N1 * 32), 256, 0, stream>>>(h1, cl1, (unsigned*)p1, N1, 5);
    pool_decode_kernel<<<blks((long)N2 * 32), 256, 0, stream>>>((unsigned*)p1, (long)N2 * 32);

    // -------- layer 2 (Cin=32 -> 64) --------
    hipMemsetAsync(acc, 0, (size_t)N2 * 800 * 4, stream);
    hipMemsetAsync(deg, 0, (size_t)N2 * 4, stream);
    scatter_edges_kernel<<<blks((long)E2 << 5), 256, 0, stream>>>(ei2, ps2, p1, acc, deg, E2, 5, 800);
    normalize_acc_kernel<<<N2, 256, 0, stream>>>(acc, deg, 800);
    wmma_gemm_kernel<4><<<blks((long)(N2 / 16) * 32), 256, 0, stream>>>(
        acc, 800, 800, w2f, p1, 32, 32, r2f, nullptr, nullptr, b2, h2, N2, 1);
    fill_u32_kernel<<<blks((long)N3 * 64), 256, 0, stream>>>((unsigned*)p2, ENC_NEGINF, (long)N3 * 64);
    pool_scatter_kernel<<<blks((long)N2 * 64), 256, 0, stream>>>(h2, cl2, (unsigned*)p2, N2, 6);
    pool_decode_kernel<<<blks((long)N3 * 64), 256, 0, stream>>>((unsigned*)p2, (long)N3 * 64);

    // -------- layer 3 (Cin=64 -> 128) --------
    hipMemsetAsync(acc, 0, (size_t)N3 * 1600 * 4, stream);
    hipMemsetAsync(deg, 0, (size_t)N3 * 4, stream);
    scatter_edges_kernel<<<blks((long)E3 << 6), 256, 0, stream>>>(ei3, ps3, p2, acc, deg, E3, 6, 1600);
    normalize_acc_kernel<<<N3, 256, 0, stream>>>(acc, deg, 1600);
    wmma_gemm_kernel<8><<<blks((long)(N3 / 16) * 32), 256, 0, stream>>>(
        acc, 1600, 1600, w3f, p2, 64, 64, r3f, nullptr, nullptr, b3, h3, N3, 1);
    fill_u32_kernel<<<blks((long)SEGS * 128), 256, 0, stream>>>((unsigned*)p3, ENC_NEGINF, (long)SEGS * 128);
    pool_scatter_kernel<<<blks((long)N3 * 128), 256, 0, stream>>>(h3, cl3, (unsigned*)p3, N3, 7);
    pool_decode_kernel<<<blks((long)SEGS * 128), 256, 0, stream>>>((unsigned*)p3, (long)SEGS * 128);

    // -------- head: fc1 (WMMA) + fc2 + log_softmax --------
    wmma_gemm_kernel<8><<<blks((long)(4096 / 16) * 32), 256, 0, stream>>>(
        p3, 512, 512, fcf, nullptr, 0, 0, nullptr, nullptr, nullptr, fc1b, z1, 4096, 1);
    head_kernel<<<blks(4096), 256, 0, stream>>>(z1, fc2w, fc2b, (float*)d_out);
}
